// AdaConv2d_28647431864926
// MI455X (gfx1250) — matrix operand order repaired and checked by reading, not compiled, not run on hardware
//
#include <hip/hip_runtime.h>

typedef __attribute__((ext_vector_type(16))) __bf16 v16bf;
typedef __attribute__((ext_vector_type(8)))  float  v8f;
typedef int v4i_t __attribute__((vector_size(16)));
typedef __attribute__((address_space(1))) v4i_t* gv4i;   // global int4*
typedef __attribute__((address_space(3))) v4i_t* lv4i;   // LDS int4*

#define EPS_  1e-5f
#define B_    4
#define C_    512
#define H_    128
#define W_    128
#define HWTOT (H_ * W_)
#define PH_   130
#define PW_   130
#define CIB   32      // ci per K-block in WMMA conv
#define TM    64      // co tile per block
#define TN    128     // w tile per block (full row)
#define TP2   16      // spatial positions per block in adaconv
#define WSTRIDE 288   // halves per co in W LDS tile (9 taps * 32 ci)

#if defined(__HIP_DEVICE_COMPILE__) && __has_builtin(__builtin_amdgcn_global_load_async_to_lds_b128)
#define ASYNC_LDS 1
#endif

struct __align__(16) U128 { unsigned int x[4]; };
union Frag { U128 q[2]; v16bf v; };

__device__ __forceinline__ unsigned short f2bf(float f) {
  unsigned int u = __float_as_uint(f);
  u += 0x7FFFu + ((u >> 16) & 1u);   // round-to-nearest-even
  return (unsigned short)(u >> 16);
}

__device__ __forceinline__ int refl(int v) {
  v = v < 0 ? -v : v;
  return v > (H_ - 1) ? (2 * (H_ - 1) - v) : v;
}

// ---------------- K1: instance-norm statistics (mean, rstd) per (b,c) ----------------
__global__ void k_in_stats(const float* __restrict__ x, float* __restrict__ stats) {
  __shared__ float s1[256], s2[256];
  const int bc = blockIdx.x;
  const float* p = x + (size_t)bc * HWTOT;
  float s = 0.f, q = 0.f;
  for (int k = threadIdx.x; k < HWTOT; k += 256) {
    float v = p[k];
    s += v; q += v * v;
  }
  s1[threadIdx.x] = s; s2[threadIdx.x] = q;
  __syncthreads();
  for (int off = 128; off > 0; off >>= 1) {
    if ((int)threadIdx.x < off) {
      s1[threadIdx.x] += s1[threadIdx.x + off];
      s2[threadIdx.x] += s2[threadIdx.x + off];
    }
    __syncthreads();
  }
  if (threadIdx.x == 0) {
    float mean = s1[0] * (1.0f / HWTOT);
    float var  = s2[0] * (1.0f / HWTOT) - mean * mean;
    stats[2 * bc]     = mean;
    stats[2 * bc + 1] = rsqrtf(var + EPS_);
  }
}

// ---- K2: fused norm + grouped-spatial + pointwise + bias, writes padded NHWC bf16 ----
__global__ void k_adaconv(const float* __restrict__ x,  const float* __restrict__ wsp,
                          const float* __restrict__ wpw, const float* __restrict__ bias,
                          const float* __restrict__ stats, unsigned short* __restrict__ ysp) {
  const int ch = threadIdx.x;           // 0..511
  const int b  = ch >> 7, g = ch & 127;

  float wpv[4];
#pragma unroll
  for (int m = 0; m < 4; m++) wpv[m] = wpw[((b * 128 + g) << 2) + m];

  float ew[4][9];
#pragma unroll
  for (int i = 0; i < 4; i++) {
#pragma unroll
    for (int t = 0; t < 9; t++) {
      float s = 0.f;
#pragma unroll
      for (int m = 0; m < 4; m++)
        s += wpv[m] * wsp[(((b * 512 + 4 * g + m) * 4 + i) * 9) + t];
      ew[i][t] = s;
    }
  }
  float mu[4], rs[4];
#pragma unroll
  for (int i = 0; i < 4; i++) {
    mu[i] = stats[2 * (b * 512 + 4 * g + i)];
    rs[i] = stats[2 * (b * 512 + 4 * g + i) + 1];
  }
  const float bv = bias[b * 128 + g];

  for (int j = 0; j < TP2; j++) {
    int p = blockIdx.x * TP2 + j;
    int h = p >> 7, w = p & 127;
    float acc = bv;
#pragma unroll
    for (int i = 0; i < 4; i++) {
      const float* xc = x + (size_t)(b * 512 + 4 * g + i) * HWTOT;
#pragma unroll
      for (int kh = 0; kh < 3; kh++) {
        int hr = refl(h - 1 + kh);
#pragma unroll
        for (int kw = 0; kw < 3; kw++) {
          int wc = refl(w - 1 + kw);
          acc += ew[i][kh * 3 + kw] * ((xc[hr * W_ + wc] - mu[i]) * rs[i]);
        }
      }
    }
    ysp[((size_t)((h + 1) * PW_ + (w + 1))) * C_ + ch] = f2bf(acc);
  }
}

// ---------------- K2b/K2c: reflect-pad border fill (rows, then cols) ----------------
__global__ void k_pad_rows(unsigned short* __restrict__ ysp) {
  int t = blockIdx.x * 256 + threadIdx.x;
  if (t >= 2 * PW_ * C_) return;
  int ch = t & (C_ - 1);
  int wp = (t >> 9) % PW_;
  int which = t / (PW_ * C_);
  int dst_h = which ? (PH_ - 1) : 0;
  int src_h = which ? (PH_ - 3) : 2;
  ysp[((size_t)(dst_h * PW_ + wp)) * C_ + ch] = ysp[((size_t)(src_h * PW_ + wp)) * C_ + ch];
}
__global__ void k_pad_cols(unsigned short* __restrict__ ysp) {
  int t = blockIdx.x * 256 + threadIdx.x;
  if (t >= 2 * PH_ * C_) return;
  int ch = t & (C_ - 1);
  int hp = (t >> 9) % PH_;
  int which = t / (PH_ * C_);
  int dst_w = which ? (PW_ - 1) : 0;
  int src_w = which ? (PW_ - 3) : 2;
  ysp[((size_t)(hp * PW_ + dst_w)) * C_ + ch] = ysp[((size_t)(hp * PW_ + src_w)) * C_ + ch];
}

// ---------------- K2d: conv_w f32 [co][ci][3][3] -> bf16 [co][tap][ci] ----------------
__global__ void k_wconv_bf16(const float* __restrict__ cw, unsigned short* __restrict__ wb) {
  int t = blockIdx.x * 256 + threadIdx.x;
  if (t >= C_ * 9 * C_) return;
  int ci  = t & 511;
  int tap = (t >> 9) % 9;
  int co  = t / (9 * 512);
  wb[t] = f2bf(cw[(size_t)(co * 512 + ci) * 9 + tap]);   // t == (co*9+tap)*512+ci
}

// ---------------- K3: implicit-GEMM 3x3 conv via bf16 WMMA, f32 accumulate ----------------
// block: 256 threads (8 waves). Tile: 64 co x 128 w at one h row.
// waves: wm = wid&3 -> 16-co subtile; wn = wid>>2 -> 64-w half (4 WMMA N-subtiles each).
// LDS chunk indices XOR-swizzled to kill the 4-way bank conflict on fragment reads.
__launch_bounds__(256)
__global__ void k_conv_wmma(const unsigned short* __restrict__ ysp,
                            const unsigned short* __restrict__ wb,
                            const float* __restrict__ cbias,
                            float* __restrict__ out) {
  __shared__ __align__(16) unsigned short Yl[3 * 130 * CIB];   // 24960 B  [row][col][ci] (swizzled)
  __shared__ __align__(16) unsigned short Wl[TM * WSTRIDE];    // 36864 B  [co][tap][ci] (swizzled)

  const int tid  = threadIdx.x;
  const int lane = tid & 31, wid = tid >> 5;
  const int wm = wid & 3, wn = wid >> 2;
  const int h   = blockIdx.x;
  const int co0 = blockIdx.y * TM;

  v8f acc[4] = {};

  for (int cbk = 0; cbk < C_; cbk += CIB) {
    __syncthreads();
    // stage Y tile: padded rows h..h+2, full padded width 0..129, 32 ci
    for (int k = tid; k < 1560; k += 256) {            // 1560 x 16B
      int pix = k >> 2, chunk = k & 3;
      int row = pix / 130, col = pix - row * 130;
      size_t g = ((size_t)((h + row) * PW_ + col)) * C_ + cbk + chunk * 8;
      int dst = pix * CIB + (((chunk ^ (pix >> 2)) & 3) << 3);
#ifdef ASYNC_LDS
      __builtin_amdgcn_global_load_async_to_lds_b128(
          (gv4i)(ysp + g), (lv4i)&Yl[dst], 0, 0);
#else
      *(U128*)&Yl[dst] = *(const U128*)(ysp + g);
#endif
    }
    // stage W tile: 64 co x 9 taps x 32 ci
    for (int k = tid; k < 2304; k += 256) {            // 2304 x 16B
      int col_ = k / 36;
      int rem  = k - col_ * 36;
      int tap = rem >> 2, chunk = rem & 3;
      size_t g = ((size_t)((co0 + col_) * 9 + tap)) * C_ + cbk + chunk * 8;
      int dst = col_ * WSTRIDE + tap * CIB + (((chunk ^ (col_ >> 2)) & 3) << 3);
#ifdef ASYNC_LDS
      __builtin_amdgcn_global_load_async_to_lds_b128(
          (gv4i)(wb + g), (lv4i)&Wl[dst], 0, 0);
#else
      *(U128*)&Wl[dst] = *(const U128*)(wb + g);
#endif
    }
    if (cbk + CIB < C_) {   // warm next K-block in L2 (global_prefetch_b8)
      size_t gy = ((size_t)(h * PW_ + (tid & 127))) * C_ + cbk + CIB;
      __builtin_prefetch(ysp + gy, 0, 1);
      size_t gw = ((size_t)((co0 + (tid & 63)) * 9)) * C_ + cbk + CIB;
      __builtin_prefetch(wb + gw, 0, 1);
    }
#ifdef ASYNC_LDS
#if __has_builtin(__builtin_amdgcn_s_wait_asynccnt)
    __builtin_amdgcn_s_wait_asynccnt(0);
#else
    asm volatile("s_wait_asynccnt 0x0" ::: "memory");
#endif
#endif
    __syncthreads();

#pragma unroll
    for (int t = 0; t < 9; t++) {
      const int row = t / 3, ck = t - row * 3;
      // A fragment (16co x 32ci): lanes 0-15 K0-7/K16-23; lanes 16-31 K8-15/K24-31
      Frag A;
      {
        int colane = wm * 16 + (lane & 15);
        int c0 = (lane & 16) ? 1 : 0;
        int sw = (colane >> 2) & 3;
        int base = colane * WSTRIDE + t * CIB;
        A.q[0] = *(const U128*)&Wl[base + ((c0 ^ sw) << 3)];
        A.q[1] = *(const U128*)&Wl[base + (((c0 + 2) ^ sw) << 3)];
      }
#pragma unroll
      for (int s = 0; s < 4; s++) {
        // B fragment (32ci x 16n): lanes 0-15 K0-7/K16-23; lanes 16-31 K8-15/K24-31
        Frag Bf;
        int nloc = wn * 64 + s * 16 + (lane & 15);
        int pix = row * 130 + ck + nloc;
        int c0 = (lane & 16) ? 1 : 0;
        int swy = (pix >> 2) & 3;
        int base = pix * CIB;
        Bf.q[0] = *(const U128*)&Yl[base + ((c0 ^ swy) << 3)];
        Bf.q[1] = *(const U128*)&Yl[base + (((c0 + 2) ^ swy) << 3)];
        acc[s] = __builtin_amdgcn_wmma_f32_16x16x32_bf16(
            false, A.v, false, Bf.v, (short)0, acc[s], false, false);
      }
    }
  }

  // epilogue: C/D layout -> NCHW f32 + bias
#pragma unroll
  for (int s = 0; s < 4; s++) {
    int wglob = wn * 64 + s * 16 + (lane & 15);
#pragma unroll
    for (int r = 0; r < 8; r++) {
      int m  = (lane & 16) ? (8 + r) : r;
      int co = co0 + wm * 16 + m;
      out[(size_t)co * HWTOT + h * W_ + wglob] = acc[s][r] + cbias[co];
    }
  }
}

extern "C" void kernel_launch(void* const* d_in, const int* in_sizes, int n_in,
                              void* d_out, int out_size, void* d_ws, size_t ws_size,
                              hipStream_t stream) {
  (void)in_sizes; (void)n_in; (void)out_size; (void)ws_size;
  const float* x    = (const float*)d_in[0];
  const float* wsp  = (const float*)d_in[1];
  const float* wpw  = (const float*)d_in[2];
  const float* bias = (const float*)d_in[3];
  const float* cw   = (const float*)d_in[4];
  const float* cb   = (const float*)d_in[5];
  float* out = (float*)d_out;

  char* ws = (char*)d_ws;
  float* stats        = (float*)ws;                                     // 16 KB
  unsigned short* ysp = (unsigned short*)(ws + (1 << 14));              // 130*130*512*2 = 17.3 MB
  unsigned short* wb  = (unsigned short*)(ws + (1 << 14) + (size_t)PH_ * PW_ * C_ * 2); // 4.7 MB

  k_in_stats <<<B_ * C_,                  256, 0, stream>>>(x, stats);
  k_adaconv  <<<HWTOT / TP2,              512, 0, stream>>>(x, wsp, wpw, bias, stats, ysp);
  k_pad_rows <<<(2 * PW_ * C_ + 255)/256, 256, 0, stream>>>(ysp);
  k_pad_cols <<<(2 * PH_ * C_ + 255)/256, 256, 0, stream>>>(ysp);
  k_wconv_bf16<<<(C_ * 9 * C_) / 256,     256, 0, stream>>>(cw, wb);
  k_conv_wmma<<<dim3(H_, C_ / TM),        256, 0, stream>>>(ysp, wb, cb, out);
}
